// SubBandMultiScaleSpectrogramLoss_20298015441387
// MI455X (gfx1250) — compile-verified, matrix operand order checked by reading, not run on hardware
//
#include <hip/hip_runtime.h>
#include <hip/hip_bf16.h>
#include <math.h>

typedef __attribute__((ext_vector_type(16))) _Float16 v16h;
typedef __attribute__((ext_vector_type(8)))  _Float16 v8h;
typedef __attribute__((ext_vector_type(8)))  float    v8f;

#define NSAMP   72000
#define NBATCH  16
#define NMELS   64
#define SPAD    82304           // f16 elems per (signal,batch): 72000 + 5*2048, rounded to mult of 128
#define FSTRIDE 1056            // floats per stats array (F max = 1025)

// ---------------------------------------------------------------------------
// Kernel 1: reflect-pad + f16 convert.  xp[(sig*16+b)*SPAD + j]
// j in [0, 72000+n_fft) -> reflect-padded sample; j in [72000+n_fft, npTotal) -> 0 slack
// ---------------------------------------------------------------------------
__global__ void msl_pad_k(const float* __restrict__ xh, const float* __restrict__ x,
                          _Float16* __restrict__ xp, int pad, int npTotal, int npValid) {
    int j = blockIdx.x * 256 + threadIdx.x;
    if (j >= npTotal) return;
    int b   = blockIdx.y;
    int sig = blockIdx.z;
    const float* src = (sig == 0) ? xh : x;
    float v = 0.0f;
    if (j < npValid) {
        int s = j - pad;
        if (s < 0) s = -s;
        else if (s >= NSAMP) s = 2 * NSAMP - 2 - s;
        v = src[b * NSAMP + s];
    }
    xp[((size_t)(sig * NBATCH + b)) * SPAD + j] = (_Float16)v;
}

// ---------------------------------------------------------------------------
// Kernel 2: windowed DFT basis in WMMA B-operand lane layout.
// basis[((ft*nch + c)*32 + lane)*32 + {0..15 cos | 16..31 sin}]
// B layout (32x16 f16): lane&15 = N, lanes<16 hold K {0-7,16-23}, lanes>=16 hold K {8-15,24-31}
// Hann window w[k] folded in (window depends only on contraction dim k).
// ---------------------------------------------------------------------------
__global__ void msl_basis_k(_Float16* __restrict__ basis, int n_fft, int nch) {
    int lane = threadIdx.x;           // 32 threads
    int c    = blockIdx.x;            // k-chunk
    int ft   = blockIdx.y;            // freq tile
    int n    = ft * 16 + (lane & 15);
    int kb   = (lane < 16) ? 0 : 8;
    float inv = 2.0f / (float)n_fft;  // angle in units of pi: 2*k*n/n_fft
    v16h hc, hs;
#pragma unroll
    for (int j = 0; j < 16; ++j) {
        int k = c * 32 + ((j < 8) ? (kb + j) : (16 + kb + (j - 8)));
        float w = 0.5f - 0.5f * cospif((float)k * inv);   // periodic hann
        int kn = (k * n) & (n_fft - 1);
        float a = (float)kn * inv;
        hc[j] = (_Float16)(w * cospif(a));
        hs[j] = (_Float16)(w * sinpif(a));                // sign irrelevant for magnitude
    }
    size_t off = (((size_t)ft * nch + c) * 32 + lane) * 32;
    *(v16h*)(basis + off)      = hc;
    *(v16h*)(basis + off + 16) = hs;
}

// ---------------------------------------------------------------------------
// Kernel 3: WMMA DFT-GEMM + magnitude diff + per-frequency reduction.
// One wave per (16 frames x 16 freqs) tile per batch. 4 f32 accumulators
// (re/im for x_hat and x) share the cos/sin B operands each K-chunk.
// A layout (16x32 f16): lane&15 = M (frame row), same K striping as B.
// ---------------------------------------------------------------------------
__global__ void msl_gemm_k(const _Float16* __restrict__ xp, const _Float16* __restrict__ basis,
                           float* __restrict__ s1a, float* __restrict__ s2a,
                           int hop, int nch, int F, float invnorm) {
    int lane = threadIdx.x;
    int ft = blockIdx.x, mt = blockIdx.y, b = blockIdx.z;
    int m  = mt * 16 + (lane & 15);            // frame index this lane loads (A row)
    int kb = (lane < 16) ? 0 : 8;

    const _Float16* pxh = xp + (size_t)b * SPAD + (size_t)m * hop;
    const _Float16* px  = xp + (size_t)(NBATCH + b) * SPAD + (size_t)m * hop;
    const _Float16* pb  = basis + ((size_t)ft * nch * 32 + lane) * 32;

    v8f hre = {}, him = {}, xre = {}, xim = {};

    for (int c = 0; c < nch; ++c) {
        int k0 = c * 32;
        // A operands: two aligned 16B groups per signal
        v8h g0 = *(const v8h*)(pxh + k0 + kb);
        v8h g1 = *(const v8h*)(pxh + k0 + 16 + kb);
        v8h h0 = *(const v8h*)(px  + k0 + kb);
        v8h h1 = *(const v8h*)(px  + k0 + 16 + kb);
        v16h ah, ax;
#pragma unroll
        for (int i = 0; i < 8; ++i) {
            ah[i] = g0[i]; ah[8 + i] = g1[i];
            ax[i] = h0[i]; ax[8 + i] = h1[i];
        }
        // B operands: cos / sin basis, contiguous 32B each per lane
        const v16h bc = *(const v16h*)(pb + (size_t)c * 1024);
        const v16h bs = *(const v16h*)(pb + (size_t)c * 1024 + 16);
        // prefetch next chunk's basis line (global_prefetch_b8)
        __builtin_prefetch(pb + (size_t)(c + 1) * 1024, 0, 0);

        hre = __builtin_amdgcn_wmma_f32_16x16x32_f16(false, ah, false, bc, (short)0, hre, false, false);
        him = __builtin_amdgcn_wmma_f32_16x16x32_f16(false, ah, false, bs, (short)0, him, false, false);
        xre = __builtin_amdgcn_wmma_f32_16x16x32_f16(false, ax, false, bc, (short)0, xre, false, false);
        xim = __builtin_amdgcn_wmma_f32_16x16x32_f16(false, ax, false, bs, (short)0, xim, false, false);
    }

    // C/D layout: lane&15 = N (freq), VGPR r -> M = r (+8 for lanes>=16).
    float s1 = 0.0f, s2 = 0.0f;
#pragma unroll
    for (int r = 0; r < 8; ++r) {
        float mh = sqrtf(hre[r] * hre[r] + him[r] * him[r]) * invnorm;
        float mx = sqrtf(xre[r] * xre[r] + xim[r] * xim[r]) * invnorm;
        float d  = mh - mx;
        s1 += fabsf(d);
        s2 += d * d;
    }
    // fold the two half-waves (M rows 0-7 with 8-15) -> lanes 0-15 hold per-N sums
    s1 += __shfl_xor(s1, 16, 32);
    s2 += __shfl_xor(s2, 16, 32);

    int f = ft * 16 + (lane & 15);
    if (lane < 16 && f < F) {
        atomicAdd(&s1a[f], s1);
        atomicAdd(&s2a[f], s2);
    }
}

// ---------------------------------------------------------------------------
// Kernel 4: mel-band slicing (replicates melscale_fbanks support extraction,
// consecutive-duplicate merge) + final loss reduction. One block, 64 threads.
// ---------------------------------------------------------------------------
__global__ void msl_final_k(const float* __restrict__ stats, float* __restrict__ out) {
    __shared__ int f0s[NMELS], f1s[NMELS];
    __shared__ float total;
    if (threadIdx.x == 0) total = 0.0f;
    __syncthreads();

    const float melMax = 2595.0f * log10f(1.0f + 12000.0f / 700.0f);

    for (int si = 0; si < 7; ++si) {
        int n_fft = 32 << si;
        int F = n_fft / 2 + 1;
        int hop = n_fft / 4;
        int T = 1 + NSAMP / hop;
        float df = 12000.0f / (float)(F - 1);

        int mm = threadIdx.x;              // one mel per thread
        float flo = 700.0f * (exp10f((melMax * (float)mm        / 65.0f) / 2595.0f) - 1.0f);
        float fhi = 700.0f * (exp10f((melMax * (float)(mm + 2) / 65.0f) / 2595.0f) - 1.0f);
        int f0 = -1, f1 = -1;
        for (int f = 0; f < F; ++f) {
            float fr = (float)f * df;
            if (fr > flo && fr < fhi) { if (f0 < 0) f0 = f; f1 = f + 1; }
        }
        f0s[mm] = f0; f1s[mm] = f1;
        __syncthreads();

        if (threadIdx.x == 0) {
            const float* s1 = stats + (size_t)si * 2 * FSTRIDE;
            const float* s2 = s1 + FSTRIDE;
            float acc = 0.0f; int nb = 0;
            int pf0 = -2, pf1 = -2;
            for (int q = 0; q < NMELS; ++q) {
                int a = f0s[q], bb = f1s[q];
                if (a < 0) continue;                      // empty support
                if (a == pf0 && bb == pf1) continue;      // merge consecutive dup
                pf0 = a; pf1 = bb;
                float u = 0.0f, v = 0.0f;
                for (int f = a; f < bb; ++f) { u += s1[f]; v += s2[f]; }
                float cnt = (float)(bb - a) * (float)NBATCH * (float)T;
                acc += u / cnt + sqrtf(v / cnt + 1e-8f);
                ++nb;
            }
            total += acc / (float)nb;
        }
        __syncthreads();
    }
    if (threadIdx.x == 0) out[0] = total / 7.0f;
}

// ---------------------------------------------------------------------------
extern "C" void kernel_launch(void* const* d_in, const int* in_sizes, int n_in,
                              void* d_out, int out_size, void* d_ws, size_t ws_size,
                              hipStream_t stream) {
    (void)in_sizes; (void)n_in; (void)out_size; (void)ws_size;
    const float* x_hat = (const float*)d_in[0];
    const float* x     = (const float*)d_in[1];

    char* ws = (char*)d_ws;
    // workspace layout
    const size_t statsOff  = 0;
    const size_t statsBytes = (size_t)7 * 2 * FSTRIDE * sizeof(float);       // ~59 KB
    const size_t xpOff     = 65536;
    const size_t xpBytes   = (size_t)2 * NBATCH * SPAD * sizeof(_Float16);   // ~5.27 MB
    const size_t basisOff  = xpOff + xpBytes;                                // basis reused per scale (~8.5 MB max)

    float*    stats    = (float*)(ws + statsOff);
    _Float16* xp       = (_Float16*)(ws + xpOff);
    _Float16* basisPtr = (_Float16*)(ws + basisOff);

    hipMemsetAsync(stats, 0, statsBytes, stream);

    for (int si = 0; si < 7; ++si) {
        int n_fft  = 1 << (5 + si);             // 32 .. 2048
        int hop    = n_fft / 4;
        int F      = n_fft / 2 + 1;
        int T      = 1 + NSAMP / hop;
        int npValid = NSAMP + n_fft;            // reflect-padded length
        int npTotal = NSAMP + 5 * n_fft;        // + zero slack for ragged frame tiles
        int ftiles = (F + 15) / 16;
        int nch    = n_fft / 32;
        float invnorm = 1.0f / sqrtf(0.375f * (float)n_fft);   // 1/sqrt(sum(hann^2)) = 1/sqrt(3n/8)

        msl_pad_k<<<dim3((npTotal + 255) / 256, NBATCH, 2), 256, 0, stream>>>(
            x_hat, x, xp, n_fft / 2, npTotal, npValid);

        msl_basis_k<<<dim3(nch, ftiles), 32, 0, stream>>>(basisPtr, n_fft, nch);

        msl_gemm_k<<<dim3(ftiles, (T + 15) / 16, NBATCH), 32, 0, stream>>>(
            xp, basisPtr,
            stats + (size_t)si * 2 * FSTRIDE,
            stats + (size_t)si * 2 * FSTRIDE + FSTRIDE,
            hop, nch, F, invnorm);
    }

    msl_final_k<<<1, NMELS, 0, stream>>>(stats, (float*)d_out);
}
